// GVPPartModel_51651276702442
// MI455X (gfx1250) — matrix operand order repaired and checked by reading, not compiled, not run on hardware
//
#include <hip/hip_runtime.h>
#include <stdint.h>

// ---------------------------------------------------------------------------
// GVP-GNN forward for MI455X (gfx1250, wave32).
//  * All big GEMMs (edge messages 265/116->100, node FF 132->400 / 432->100,
//    Wout 116->100, dense1 100->200) run on v_wmma_f32_16x16x32_bf16 with f32
//    accumulation; activations are bf16-staged in LDS, weights pre-packed into
//    B-fragment order once per launch.
//  * Small 3-channel vector matmuls (wh/wv <= 33x33), layernorms, sigmoid
//    gates, residuals: f32 VALU.
//  * Segment-mean aggregation: native fire-and-forget global_atomic_add_f32
//    (inline asm, scope:SCOPE_DEV) into L2-resident tables.
// ---------------------------------------------------------------------------

#define NN 8192
#define NE 262144
#define NS 100
#define NV 16

typedef __attribute__((ext_vector_type(16))) __bf16 v16bf;
typedef __attribute__((ext_vector_type(8)))  float  v8f;

#define DEV static __device__ __forceinline__

DEV uint16_t f2bf(float f){
  union { float f; uint32_t u; } x; x.f = f;
  return (uint16_t)((x.u + 0x7fffu + ((x.u >> 16) & 1u)) >> 16); // RNE
}
DEV float sigm(float x){ return 1.f / (1.f + __expf(-x)); }
DEV float nrm3(float a, float b, float c){ return sqrtf(fmaxf(a*a + b*b + c*c, 1e-8f)); }

// Fire-and-forget fp32 atomic add at device scope (L2-side RMW, no return).
// Tracked by STOREcnt; S_ENDPGM's implicit wait-idle covers completion.
DEV void atomicAddDev(float* p, float v){
  asm volatile("global_atomic_add_f32 %0, %1, off scope:SCOPE_DEV"
               :: "v"((uint64_t)(uintptr_t)p), "v"(v) : "memory");
}

DEV v16bf make16(uint4 a, uint4 b){
  union { uint32_t u[8]; v16bf v; } r;
  r.u[0]=a.x; r.u[1]=a.y; r.u[2]=a.z; r.u[3]=a.w;
  r.u[4]=b.x; r.u[5]=b.y; r.u[6]=b.z; r.u[7]=b.w;
  return r.v;
}
// 16-bit A fragment (16xK row-major bf16 in LDS): lane<16 -> K{0..7,16..23},
// lane>=16 -> K{8..15,24..31} of row (lane&15), per CDNA5 ISA A layout.
DEV v16bf load_afrag(const uint16_t* X, int ldx, int kt, int lane){
  int row = lane & 15;
  int kb  = kt*32 + ((lane >> 4) << 3);
  const uint4* p = (const uint4*)(X + row*ldx + kb);
  const uint4* q = (const uint4*)(X + row*ldx + kb + 16);
  return make16(*p, *q);
}
// B fragments pre-packed: [kt][nt][lane][8 dwords]
DEV v16bf load_bfrag(const uint32_t* B, int NT, int kt, int nt, int lane){
  const uint4* p = (const uint4*)(B + (((size_t)(kt*NT + nt)*32 + lane) << 3));
  return make16(p[0], p[1]);
}
template<int KT>
DEV void load_afrags(const uint16_t* X, int ldx, int lane, v16bf* a){
#pragma unroll
  for (int kt=0; kt<KT; ++kt) a[kt] = load_afrag(X, ldx, kt, lane);
}
template<int KT>
DEV v8f wmma_rowtile(const v16bf* a, const uint32_t* B, int NT, int nt, int lane){
  v8f acc = {0.f,0.f,0.f,0.f,0.f,0.f,0.f,0.f};
#pragma unroll
  for (int kt=0; kt<KT; ++kt){
    v16bf b = load_bfrag(B, NT, kt, nt, lane);
    acc = __builtin_amdgcn_wmma_f32_16x16x32_bf16(false, a[kt], false, b,
                                                  (short)0, acc, false, false);
  }
  return acc;
}
// D layout: VGPR r, lanes 0-15 -> (M=r, N=lane); lanes 16-31 -> (M=r+8, N=lane-16)
DEV void store_tile_bf16(v8f acc, const float* bias, int nt, int Nreal, int lane,
                         uint16_t* Y, int ldy, bool relu){
  int nl = lane & 15, n = nt*16 + nl, mb = (lane >> 4) * 8;
  if (n < Nreal){
    float bv = bias[n];
#pragma unroll
    for (int r=0;r<8;++r){
      float val = acc[r] + bv;
      if (relu) val = fmaxf(val, 0.f);
      Y[(mb + r)*ldy + n] = f2bf(val);
    }
  }
}

// ---------------------------------------------------------------------------
// Weight packing into B-fragment order: W is [N][K] row-major (so B[k][n]=W[n][k]).
// lane<16 covers K {0..7,16..23} of a kt (2 bf16/dword), lane>=16 covers K {8..15,24..31}.
__global__ void k_pack(const float* W, int N, int K, int NT, uint32_t* out){
  int frag = blockIdx.x;
  int kt = frag / NT, nt = frag - kt*NT;
  int l = threadIdx.x;
  int n = nt*16 + (l & 15);
  int kbase = kt*32 + ((l >> 4) << 3);
  uint32_t* dst = out + (((size_t)frag*32 + l) << 3);
  for (int j=0;j<8;++j){
    int k0 = kbase + 2*j + ((j >> 2) << 3), k1 = k0 + 1;
    float f0 = (n < N && k0 < K) ? W[(size_t)n*K + k0] : 0.f;
    float f1 = (n < N && k1 < K) ? W[(size_t)n*K + k1] : 0.f;
    dst[j] = (uint32_t)f2bf(f0) | ((uint32_t)f2bf(f1) << 16);
  }
}

__global__ void k_zero(float* p, int n){
  int i = blockIdx.x*256 + threadIdx.x;
  if (i < n) p[i] = 0.f;
}

// ---------------------------------------------------------------------------
// Node embedding: Wv_ln + Wv_gvp (tiny, pure VALU). 1 wave = 16 nodes.
struct EmbNodeP {
  const float *hs, *hv, *g, *b, *wh, *ws, *wsb, *wv;
  float *s, *v;
};
__global__ void __launch_bounds__(32) k_node_embed(EmbNodeP P){
  __shared__ float slnS[16][6];
  __shared__ float vhS[16][16][3];
  __shared__ float vnS[16][16];
  __shared__ float vlnS[16][3][3];
  int lane = threadIdx.x, n0 = blockIdx.x*16;
  if (lane < 16){
    int node = n0 + lane;
    float x[6], mu = 0.f;
    for (int i=0;i<6;++i){ x[i]=P.hs[node*6+i]; mu+=x[i]; }
    mu *= (1.f/6.f);
    float var=0.f; for (int i=0;i<6;++i){ float d=x[i]-mu; var+=d*d; }
    float rs = rsqrtf(var*(1.f/6.f) + 1e-5f);
    for (int i=0;i<6;++i) slnS[lane][i] = (x[i]-mu)*rs*P.g[i] + P.b[i];
    float vv[3][3], acc=0.f;
    for (int k=0;k<3;++k){
      float ss=0.f;
      for (int c=0;c<3;++c){ vv[k][c]=P.hv[(node*3+k)*3+c]; ss+=vv[k][c]*vv[k][c]; }
      acc += fmaxf(ss, 1e-8f);
    }
    float rn = rsqrtf(acc*(1.f/3.f));
    for (int k=0;k<3;++k) for (int c=0;c<3;++c) vlnS[lane][k][c] = vv[k][c]*rn;
  }
  // vh = wh[16][3] @ vln ; vn
  for (int idx=lane; idx<16*16; idx+=32){
    int m=idx>>4, h=idx&15;
    float a0=0,a1=0,a2=0;
    for (int k=0;k<3;++k){
      float w=P.wh[h*3+k];
      a0+=w*vlnS[m][k][0]; a1+=w*vlnS[m][k][1]; a2+=w*vlnS[m][k][2];
    }
    vhS[m][h][0]=a0; vhS[m][h][1]=a1; vhS[m][h][2]=a2;
    vnS[m][h]=nrm3(a0,a1,a2);
  }
  for (int idx=lane; idx<16*100; idx+=32){
    int m=idx/100, j=idx-m*100;
    float a = P.wsb[j];
    for (int i=0;i<6;++i)  a += P.ws[j*22+i]*slnS[m][i];
    for (int h=0;h<16;++h) a += P.ws[j*22+6+h]*vnS[m][h];
    P.s[(size_t)(n0+m)*NS + j] = a;
  }
  for (int idx=lane; idx<16*48; idx+=32){
    int m=idx/48, r=idx-m*48, o=r/3, c=r-o*3;
    float a=0.f;
    for (int h=0;h<16;++h) a += P.wv[o*16+h]*vhS[m][h][c];
    P.v[((size_t)(n0+m)*NV + o)*3 + c] = a;
  }
}

// ---------------------------------------------------------------------------
// Edge embedding: We_ln + We_gvp. 1 wave = 16 edges.
struct EmbEdgeP {
  const float *hes, *hev, *g, *b, *wh, *ws, *wsb, *wv;
  float *es, *ev;
};
__global__ void __launch_bounds__(32) k_edge_embed(EmbEdgeP P){
  __shared__ float esS[16][33];
  __shared__ float muE[16], rsE[16];
  int lane = threadIdx.x, e0 = blockIdx.x*16;
  if (lane < 16){
    size_t e = e0 + lane;
    float mu=0.f; for (int i=0;i<32;++i) mu += P.hes[e*32+i];
    mu *= (1.f/32.f);
    float var=0.f; for (int i=0;i<32;++i){ float d=P.hes[e*32+i]-mu; var+=d*d; }
    muE[lane]=mu; rsE[lane]=rsqrtf(var*(1.f/32.f) + 1e-5f);
  }
  for (int idx=lane; idx<16*32; idx+=32){
    int m=idx>>5, i=idx&31;
    esS[m][i] = (P.hes[(size_t)(e0+m)*32+i]-muE[m])*rsE[m]*P.g[i] + P.b[i];
  }
  if (lane < 16){
    size_t e = e0 + lane;
    float x0=P.hev[e*3], x1=P.hev[e*3+1], x2=P.hev[e*3+2];
    float rn = rsqrtf(fmaxf(x0*x0+x1*x1+x2*x2, 1e-8f));
    float whv = P.wh[0];
    float h0=whv*x0*rn, h1=whv*x1*rn, h2=whv*x2*rn;
    esS[lane][32] = nrm3(h0,h1,h2);
    float wvv = P.wv[0];
    P.ev[e*3+0]=wvv*h0; P.ev[e*3+1]=wvv*h1; P.ev[e*3+2]=wvv*h2;
  }
  for (int idx=lane; idx<16*32; idx+=32){
    int m=idx>>5, j=idx&31;
    float a = P.wsb[j];
    for (int i=0;i<33;++i) a += P.ws[j*33+i]*esS[m][i];
    P.es[(size_t)(e0+m)*32 + j] = a;
  }
}

// ---------------------------------------------------------------------------
// Message kernel: 2 waves/block, each wave owns 16 edges end-to-end.
struct MsgP {
  const float *s, *v, *es, *ev;
  const int *ei;
  const uint32_t *B0, *B1, *B2;
  const float *b0, *b1, *b2;
  const float *wh0, *wv0, *wh1, *wv1, *wh2, *wv2;
  float *dh_s, *dh_v, *cnt;
};
__global__ void __launch_bounds__(64) k_message(MsgP P){
  __shared__ __align__(16) uint16_t Xs[2][16][288];
  __shared__ __align__(16) float vhS[2][16][33][3];
  __shared__ __align__(16) float vcS[2][16][16][3];
  __shared__ __align__(16) float xvS[2][16][99];
  __shared__ int srcS[2][16], dstS[2][16];

  const int w = threadIdx.x >> 5, lane = threadIdx.x & 31;
  const int e0 = (blockIdx.x*2 + w) * 16;

  if (lane < 16){
    srcS[w][lane] = P.ei[e0 + lane];
    dstS[w][lane] = P.ei[NE + e0 + lane];
  }
  // X cols: 0..99 s[src], 100..131 es, 132..231 s[dst], 232..264 vn0, 265..287 pad0
  for (int idx=lane; idx<16*288; idx+=32){
    int e = idx/288, c = idx - e*288;
    float val = 0.f;
    if      (c < 100) val = P.s[(size_t)srcS[w][e]*NS + c];
    else if (c < 132) val = P.es[(size_t)(e0+e)*32 + (c-100)];
    else if (c < 232) val = P.s[(size_t)dstS[w][e]*NS + (c-132)];
    Xs[w][e][c] = f2bf(val);
  }
  // gather all vector inputs: xv[e][k*3+c], k: 0..15 v[src], 16 ev, 17..32 v[dst]
  for (int idx=lane; idx<16*99; idx+=32){
    int e = idx/99, r = idx - e*99, k = r/3, c = r - k*3;
    float val;
    if      (k < 16)  val = P.v[((size_t)srcS[w][e]*NV + k)*3 + c];
    else if (k == 16) val = P.ev[(size_t)(e0+e)*3 + c];
    else              val = P.v[((size_t)dstS[w][e]*NV + (k-17))*3 + c];
    xvS[w][e][r] = val;
  }
  // vh0 = wh0[33][33] @ xv ; vn0 -> X cols 232..264.
  // Each lane owns wh0 row h (loaded once into registers), sweeps all 16 edges.
  for (int h=lane; h<33; h+=32){
    float wrow[33];
#pragma unroll
    for (int k=0;k<33;++k) wrow[k] = P.wh0[h*33+k];
    for (int e=0;e<16;++e){
      float a0=0,a1=0,a2=0;
#pragma unroll
      for (int k=0;k<33;++k){
        a0 += wrow[k]*xvS[w][e][k*3+0];
        a1 += wrow[k]*xvS[w][e][k*3+1];
        a2 += wrow[k]*xvS[w][e][k*3+2];
      }
      vhS[w][e][h][0]=a0; vhS[w][e][h][1]=a1; vhS[w][e][h][2]=a2;
      Xs[w][e][232+h] = f2bf(nrm3(a0,a1,a2));
    }
  }
  // GEMM0: K=265(pad288), N=100 -> relu -> X cols 0..99 (bf16)
  {
    v16bf a[9]; load_afrags<9>(&Xs[w][0][0], 288, lane, a);
    for (int nt=0; nt<7; ++nt){
      v8f acc = wmma_rowtile<9>(a, P.B0, 7, nt, lane);
      store_tile_bf16(acc, P.b0, nt, 100, lane, &Xs[w][0][0], 288, true);
    }
  }
  for (int idx=lane; idx<16*12; idx+=32) Xs[w][idx/12][116 + idx%12] = 0; // pad for K=128
  // v1 = sigmoid(||.||) * (wv0[16][33] @ vh0); vh1 = wh1 @ v1; vn1 -> cols 100..115
  for (int idx=lane; idx<16*48; idx+=32){
    int e=idx/48, r=idx-e*48, o=r/3, c=r-o*3;
    float a=0.f;
    for (int h=0;h<33;++h) a += P.wv0[o*33+h]*vhS[w][e][h][c];
    vcS[w][e][o][c]=a;
  }
  for (int idx=lane; idx<16*16; idx+=32){
    int e=idx>>4, o=idx&15;
    float x0=vcS[w][e][o][0], x1=vcS[w][e][o][1], x2=vcS[w][e][o][2];
    float g = sigm(nrm3(x0,x1,x2));
    vcS[w][e][o][0]=x0*g; vcS[w][e][o][1]=x1*g; vcS[w][e][o][2]=x2*g;
  }
  for (int idx=lane; idx<16*16; idx+=32){
    int e=idx>>4, h=idx&15;
    float a0=0,a1=0,a2=0;
    for (int k=0;k<16;++k){
      float wt=P.wh1[h*16+k];
      a0+=wt*vcS[w][e][k][0]; a1+=wt*vcS[w][e][k][1]; a2+=wt*vcS[w][e][k][2];
    }
    vhS[w][e][h][0]=a0; vhS[w][e][h][1]=a1; vhS[w][e][h][2]=a2;
    Xs[w][e][100+h] = f2bf(nrm3(a0,a1,a2));
  }
  // GEMM1: K=116(pad128) -> relu
  {
    v16bf a[4]; load_afrags<4>(&Xs[w][0][0], 288, lane, a);
    for (int nt=0; nt<7; ++nt){
      v8f acc = wmma_rowtile<4>(a, P.B1, 7, nt, lane);
      store_tile_bf16(acc, P.b1, nt, 100, lane, &Xs[w][0][0], 288, true);
    }
  }
  // v2 gated, vh2, vn2
  for (int idx=lane; idx<16*48; idx+=32){
    int e=idx/48, r=idx-e*48, o=r/3, c=r-o*3;
    float a=0.f;
    for (int h=0;h<16;++h) a += P.wv1[o*16+h]*vhS[w][e][h][c];
    vcS[w][e][o][c]=a;
  }
  for (int idx=lane; idx<16*16; idx+=32){
    int e=idx>>4, o=idx&15;
    float x0=vcS[w][e][o][0], x1=vcS[w][e][o][1], x2=vcS[w][e][o][2];
    float g = sigm(nrm3(x0,x1,x2));
    vcS[w][e][o][0]=x0*g; vcS[w][e][o][1]=x1*g; vcS[w][e][o][2]=x2*g;
  }
  for (int idx=lane; idx<16*16; idx+=32){
    int e=idx>>4, h=idx&15;
    float a0=0,a1=0,a2=0;
    for (int k=0;k<16;++k){
      float wt=P.wh2[h*16+k];
      a0+=wt*vcS[w][e][k][0]; a1+=wt*vcS[w][e][k][1]; a2+=wt*vcS[w][e][k][2];
    }
    vhS[w][e][h][0]=a0; vhS[w][e][h][1]=a1; vhS[w][e][h][2]=a2;
    Xs[w][e][100+h] = f2bf(nrm3(a0,a1,a2));
  }
  // GEMM2 (no act) -> atomic scatter into dh_s
  {
    v16bf a[4]; load_afrags<4>(&Xs[w][0][0], 288, lane, a);
    for (int nt=0; nt<7; ++nt){
      v8f acc = wmma_rowtile<4>(a, P.B2, 7, nt, lane);
      int nl = lane & 15, n = nt*16 + nl, mb = (lane >> 4) * 8;
      if (n < 100){
        float bv = P.b2[n];
#pragma unroll
        for (int r=0;r<8;++r){
          int e = mb + r;
          atomicAddDev(&P.dh_s[(size_t)dstS[w][e]*NS + n], acc[r] + bv);
        }
      }
    }
  }
  // v3 = wv2 @ vh2 (no gate) -> atomic scatter into dh_v
  for (int idx=lane; idx<16*48; idx+=32){
    int e=idx/48, r=idx-e*48, o=r/3, c=r-o*3;
    float a=0.f;
    for (int h=0;h<16;++h) a += P.wv2[o*16+h]*vhS[w][e][h][c];
    atomicAddDev(&P.dh_v[((size_t)dstS[w][e]*NV + o)*3 + c], a);
  }
  if (lane < 16) atomicAddDev(&P.cnt[dstS[w][lane]], 1.0f);
}

// ---------------------------------------------------------------------------
// Node update: residual + norm0 + FF (2 GVPs on WMMA) + norm1. 1 wave = 16 nodes.
struct NodeP {
  float *s, *v;
  const float *dh_s, *dh_v, *cnt;
  const float *g0, *bb0, *g1, *bb1;
  const float *fwh0, *fwv0, *fwh1, *fwv1;
  const uint32_t *Bff0, *Bff1;
  const float *bff0, *bff1;
};
__global__ void __launch_bounds__(32) k_node_update(NodeP P){
  __shared__ __align__(16) float sln[16][100];
  __shared__ __align__(16) float vln[16][16][3];
  __shared__ __align__(16) float vhS[16][32][3];
  __shared__ __align__(16) float vcS[16][32][3];
  __shared__ __align__(16) uint16_t Xs[16][448];
  __shared__ float muS[16], rsS[16];

  int lane = threadIdx.x, n0 = blockIdx.x*16;

  for (int idx=lane; idx<16*100; idx+=32){
    int m=idx/100, c=idx-m*100, node=n0+m;
    float cn = fmaxf(P.cnt[node], 1.f);
    sln[m][c] = P.s[(size_t)node*NS + c] + P.dh_s[(size_t)node*NS + c] / cn;
  }
  if (lane < 16){
    float mu=0.f; for (int c=0;c<100;++c) mu += sln[lane][c];
    mu *= 0.01f;
    float var=0.f; for (int c=0;c<100;++c){ float d=sln[lane][c]-mu; var+=d*d; }
    muS[lane]=mu; rsS[lane]=rsqrtf(var*0.01f + 1e-5f);
  }
  for (int idx=lane; idx<16*100; idx+=32){
    int m=idx/100, c=idx-m*100;
    sln[m][c] = (sln[m][c]-muS[m])*rsS[m]*P.g0[c] + P.bb0[c];
  }
  for (int idx=lane; idx<16*48; idx+=32){
    int m=idx/48, r=idx-m*48, k=r/3, c=r-k*3, node=n0+m;
    float cn = fmaxf(P.cnt[node], 1.f);
    vln[m][k][c] = P.v[((size_t)node*NV+k)*3+c] + P.dh_v[((size_t)node*NV+k)*3+c]/cn;
  }
  if (lane < 16){
    float acc=0.f;
    for (int k=0;k<16;++k){
      float x0=vln[lane][k][0], x1=vln[lane][k][1], x2=vln[lane][k][2];
      acc += fmaxf(x0*x0+x1*x1+x2*x2, 1e-8f);
    }
    rsS[lane] = rsqrtf(acc*(1.f/16.f));
  }
  for (int idx=lane; idx<16*48; idx+=32){
    int m=idx/48, r=idx-m*48;
    vln[m][r/3][r%3] *= rsS[m];
  }
  // ff0: vh = fwh0[32][16] @ vln ; X = [bf16(sln), vn(32), pad] K=132->160
  for (int idx=lane; idx<16*32; idx+=32){
    int m=idx>>5, h=idx&31;
    float a0=0,a1=0,a2=0;
    for (int k=0;k<16;++k){
      float wt=P.fwh0[h*16+k];
      a0+=wt*vln[m][k][0]; a1+=wt*vln[m][k][1]; a2+=wt*vln[m][k][2];
    }
    vhS[m][h][0]=a0; vhS[m][h][1]=a1; vhS[m][h][2]=a2;
    Xs[m][100+h] = f2bf(nrm3(a0,a1,a2));
  }
  for (int idx=lane; idx<16*100; idx+=32){ int m=idx/100, c=idx-m*100; Xs[m][c]=f2bf(sln[m][c]); }
  for (int idx=lane; idx<16*28;  idx+=32){ Xs[idx/28][132 + idx%28] = 0; }
  {
    v16bf a[5]; load_afrags<5>(&Xs[0][0], 448, lane, a);
    for (int nt=0; nt<25; ++nt){
      v8f acc = wmma_rowtile<5>(a, P.Bff0, 25, nt, lane);
      store_tile_bf16(acc, P.bff0, nt, 400, lane, &Xs[0][0], 448, true); // fs -> cols 0..399
    }
  }
  // v_ff0 gated; vh1 = fwh1[32][32] @ v_ff0; vn1 -> cols 400..431; pad 432..447
  for (int idx=lane; idx<16*96; idx+=32){
    int m=idx/96, r=idx-m*96, o=r/3, c=r-o*3;
    float a=0.f;
    for (int h=0;h<32;++h) a += P.fwv0[o*32+h]*vhS[m][h][c];
    vcS[m][o][c]=a;
  }
  for (int idx=lane; idx<16*32; idx+=32){
    int m=idx>>5, o=idx&31;
    float x0=vcS[m][o][0], x1=vcS[m][o][1], x2=vcS[m][o][2];
    float g = sigm(nrm3(x0,x1,x2));
    vcS[m][o][0]=x0*g; vcS[m][o][1]=x1*g; vcS[m][o][2]=x2*g;
  }
  for (int idx=lane; idx<16*32; idx+=32){
    int m=idx>>5, h=idx&31;
    float a0=0,a1=0,a2=0;
    for (int k=0;k<32;++k){
      float wt=P.fwh1[h*32+k];
      a0+=wt*vcS[m][k][0]; a1+=wt*vcS[m][k][1]; a2+=wt*vcS[m][k][2];
    }
    vhS[m][h][0]=a0; vhS[m][h][1]=a1; vhS[m][h][2]=a2;
    Xs[m][400+h] = f2bf(nrm3(a0,a1,a2));
  }
  for (int idx=lane; idx<16*16; idx+=32) Xs[idx>>4][432 + (idx&15)] = 0;
  // ff1: K=432(pad448) -> s residual into sln (f32)
  {
    v16bf a[14]; load_afrags<14>(&Xs[0][0], 448, lane, a);
    for (int nt=0; nt<7; ++nt){
      v8f acc = wmma_rowtile<14>(a, P.Bff1, 7, nt, lane);
      int nl=lane&15, n=nt*16+nl, mb=(lane>>4)*8;
      if (n < 100){
        float bv = P.bff1[n];
#pragma unroll
        for (int r=0;r<8;++r) sln[mb+r][n] += acc[r] + bv;
      }
    }
  }
  for (int idx=lane; idx<16*48; idx+=32){
    int m=idx/48, r=idx-m*48, o=r/3, c=r-o*3;
    float a=0.f;
    for (int h=0;h<32;++h) a += P.fwv1[o*32+h]*vhS[m][h][c];
    vln[m][o][c] += a;
  }
  // norm1 -> write back
  if (lane < 16){
    float mu=0.f; for (int c=0;c<100;++c) mu += sln[lane][c];
    mu *= 0.01f;
    float var=0.f; for (int c=0;c<100;++c){ float d=sln[lane][c]-mu; var+=d*d; }
    muS[lane]=mu; rsS[lane]=rsqrtf(var*0.01f + 1e-5f);
  }
  for (int idx=lane; idx<16*100; idx+=32){
    int m=idx/100, c=idx-m*100;
    P.s[(size_t)(n0+m)*NS + c] = (sln[m][c]-muS[m])*rsS[m]*P.g1[c] + P.bb1[c];
  }
  if (lane < 16){
    float acc=0.f;
    for (int k=0;k<16;++k){
      float x0=vln[lane][k][0], x1=vln[lane][k][1], x2=vln[lane][k][2];
      acc += fmaxf(x0*x0+x1*x1+x2*x2, 1e-8f);
    }
    rsS[lane] = rsqrtf(acc*(1.f/16.f));
  }
  for (int idx=lane; idx<16*48; idx+=32){
    int m=idx/48, r=idx-m*48;
    P.v[((size_t)(n0+m)*NV + r/3)*3 + (r%3)] = vln[m][r/3][r%3] * rsS[m];
  }
}

// ---------------------------------------------------------------------------
// Output head: Wout_ln + Wout_gvp (WMMA) + dense1 (WMMA, relu) + dense2 dot.
struct OutP {
  const float *s, *v, *gO, *bO, *whO, *wsbO;
  const uint32_t *BO, *Bd1;
  const float *bd1, *wd2, *bd2;
  float *o;
};
__global__ void __launch_bounds__(32) k_out(OutP P){
  __shared__ __align__(16) float sln[16][100];
  __shared__ __align__(16) float vln[16][16][3];
  __shared__ __align__(16) uint16_t Xs[16][128];
  __shared__ __align__(16) float hS[16][200];
  __shared__ float muS[16], rsS[16];
  int lane = threadIdx.x, n0 = blockIdx.x*16;

  for (int idx=lane; idx<16*100; idx+=32){
    int m=idx/100, c=idx-m*100;
    sln[m][c] = P.s[(size_t)(n0+m)*NS + c];
  }
  if (lane < 16){
    float mu=0.f; for (int c=0;c<100;++c) mu += sln[lane][c];
    mu *= 0.01f;
    float var=0.f; for (int c=0;c<100;++c){ float d=sln[lane][c]-mu; var+=d*d; }
    muS[lane]=mu; rsS[lane]=rsqrtf(var*0.01f + 1e-5f);
  }
  for (int idx=lane; idx<16*100; idx+=32){
    int m=idx/100, c=idx-m*100;
    sln[m][c] = (sln[m][c]-muS[m])*rsS[m]*P.gO[c] + P.bO[c];
    Xs[m][c] = f2bf(sln[m][c]);
  }
  for (int idx=lane; idx<16*48; idx+=32){
    int m=idx/48, r=idx-m*48;
    vln[m][r/3][r%3] = P.v[((size_t)(n0+m)*NV + r/3)*3 + (r%3)];
  }
  if (lane < 16){
    float acc=0.f;
    for (int k=0;k<16;++k){
      float x0=vln[lane][k][0], x1=vln[lane][k][1], x2=vln[lane][k][2];
      acc += fmaxf(x0*x0+x1*x1+x2*x2, 1e-8f);
    }
    rsS[lane] = rsqrtf(acc*(1.f/16.f));
  }
  for (int idx=lane; idx<16*48; idx+=32){
    int m=idx/48, r=idx-m*48;
    vln[m][r/3][r%3] *= rsS[m];
  }
  // vn = ||whO[16][16] @ vln|| -> X cols 100..115 ; 116..127 zero
  for (int idx=lane; idx<16*16; idx+=32){
    int m=idx>>4, h=idx&15;
    float a0=0,a1=0,a2=0;
    for (int k=0;k<16;++k){
      float wt=P.whO[h*16+k];
      a0+=wt*vln[m][k][0]; a1+=wt*vln[m][k][1]; a2+=wt*vln[m][k][2];
    }
    Xs[m][100+h] = f2bf(nrm3(a0,a1,a2));
  }
  for (int idx=lane; idx<16*12; idx+=32) Xs[idx/12][116 + idx%12] = 0;
  // Wout GEMM (no act) -> X cols 0..99 ; then zero 100..127 for dense1
  {
    v16bf a[4]; load_afrags<4>(&Xs[0][0], 128, lane, a);
    for (int idx=lane; idx<16*28; idx+=32) Xs[idx/28][100 + idx%28] = 0;
    for (int nt=0; nt<7; ++nt){
      v8f acc = wmma_rowtile<4>(a, P.BO, 7, nt, lane);
      store_tile_bf16(acc, P.wsbO, nt, 100, lane, &Xs[0][0], 128, false);
    }
  }
  // dense1 (relu) -> hS (f32)
  {
    v16bf a[4]; load_afrags<4>(&Xs[0][0], 128, lane, a);
    for (int nt=0; nt<13; ++nt){
      v8f acc = wmma_rowtile<4>(a, P.Bd1, 13, nt, lane);
      int nl=lane&15, n=nt*16+nl, mb=(lane>>4)*8;
      if (n < 200){
        float bv = P.bd1[n];
#pragma unroll
        for (int r=0;r<8;++r) hS[mb+r][n] = fmaxf(acc[r]+bv, 0.f);
      }
    }
  }
  if (lane < 16){
    float a = P.bd2[0];
    for (int j=0;j<200;++j) a += hS[lane][j]*P.wd2[j];
    P.o[n0 + lane] = a;
  }
}

__global__ void k_pairsum(const float* __restrict__ o, float* __restrict__ out){
  size_t idx = (size_t)blockIdx.x*256 + threadIdx.x;
  int i = (int)(idx >> 13), j = (int)(idx & 8191);
  out[idx] = o[i] + o[j];
}

// ---------------------------------------------------------------------------
// Host side.
// Param leaves assumed in jax.tree_leaves order (sorted dict keys):
static const int kLeafSizes[93] = {
  1, 32, 1056, 1,            // We_gvp: wh, ws_b, ws_w, wv
  32, 32,                    // We_ln: b, g
  256, 100, 11600,           // Wout_gvp: wh, ws_b, ws_w
  100, 100,                  // Wout_ln: b, g
  48, 100, 2200, 256,        // Wv_gvp
  6, 6,                      // Wv_ln: b, g
  200, 20000,                // dense1: b, w
  1, 200,                    // dense2: b, w
  // 3 layers x [ff0(4), ff1(4), msg0(4), msg1(4), msg2(4), norm0(b,g), norm1(b,g)]
  512,400,52800,1024, 1024,100,43200,512, 1089,100,26500,528,
  256,100,11600,256, 256,100,11600,256, 100,100, 100,100,
  512,400,52800,1024, 1024,100,43200,512, 1089,100,26500,528,
  256,100,11600,256, 256,100,11600,256, 100,100, 100,100,
  512,400,52800,1024, 1024,100,43200,512, 1089,100,26500,528,
  256,100,11600,256, 256,100,11600,256, 100,100, 100,100,
};
enum {
  L_WE_WH=0, L_WE_WSB, L_WE_WSW, L_WE_WV, L_WELN_B, L_WELN_G,
  L_WO_WH, L_WO_WSB, L_WO_WSW, L_WOLN_B, L_WOLN_G,
  L_WV_WH, L_WV_WSB, L_WV_WSW, L_WV_WV, L_WVLN_B, L_WVLN_G,
  L_D1_B, L_D1_W, L_D2_B, L_D2_W, L_LAYER0
};

extern "C" void kernel_launch(void* const* d_in, const int* in_sizes, int n_in,
                              void* d_out, int out_size, void* d_ws, size_t ws_size,
                              hipStream_t stream){
  (void)in_sizes; (void)out_size; (void)ws_size;
  const float* hVs = (const float*)d_in[0];
  const float* hVv = (const float*)d_in[1];
  const int*   ei  = (const int*)d_in[2];
  const float* hEs = (const float*)d_in[3];
  const float* hEv = (const float*)d_in[4];

  auto leaf = [&](int idx)->const float*{
    if (n_in >= 6 + 93) return (const float*)d_in[5 + idx];
    size_t off = 0;
    for (int i=0;i<idx;++i) off += kLeafSizes[i];
    return (const float*)d_in[5] + off;
  };

  // workspace carve-out
  char* base = (char*)d_ws; size_t off = 0;
  auto carve = [&](size_t bytes)->void*{
    void* p = base + off; off = (off + bytes + 255) & ~(size_t)255; return p;
  };
  float* s    = (float*)carve((size_t)NN*NS*4);
  float* v    = (float*)carve((size_t)NN*NV*3*4);
  float* es   = (float*)carve((size_t)NE*32*4);
  float* ev   = (float*)carve((size_t)NE*3*4);
  float* dh   = (float*)carve((size_t)(NN*NS + NN*NV*3 + NN)*4);
  float* dh_s = dh;
  float* dh_v = dh + NN*NS;
  float* cnt  = dh_v + NN*NV*3;
  float* o    = (float*)carve((size_t)NN*4);
  uint32_t *B0[3], *B1[3], *B2[3], *Bff0[3], *Bff1[3];
  for (int L=0; L<3; ++L){
    B0[L]   = (uint32_t*)carve((size_t)9*7*256*4);
    B1[L]   = (uint32_t*)carve((size_t)4*7*256*4);
    B2[L]   = (uint32_t*)carve((size_t)4*7*256*4);
    Bff0[L] = (uint32_t*)carve((size_t)5*25*256*4);
    Bff1[L] = (uint32_t*)carve((size_t)14*7*256*4);
  }
  uint32_t* BO  = (uint32_t*)carve((size_t)4*7*256*4);
  uint32_t* Bd1 = (uint32_t*)carve((size_t)4*13*256*4);

  auto pack = [&](const float* W, int N, int K, int KT, int NT, uint32_t* out){
    k_pack<<<dim3(KT*NT), dim3(32), 0, stream>>>(W, N, K, NT, out);
  };
  for (int L=0; L<3; ++L){
    int b = L_LAYER0 + 24*L;
    pack(leaf(b+10), 100, 265, 9, 7,  B0[L]);    // msg0.ws_w
    pack(leaf(b+14), 100, 116, 4, 7,  B1[L]);    // msg1.ws_w
    pack(leaf(b+18), 100, 116, 4, 7,  B2[L]);    // msg2.ws_w
    pack(leaf(b+2),  400, 132, 5, 25, Bff0[L]);  // ff0.ws_w
    pack(leaf(b+6),  100, 432, 14, 7, Bff1[L]);  // ff1.ws_w
  }
  pack(leaf(L_WO_WSW), 100, 116, 4, 7,  BO);
  pack(leaf(L_D1_W),   200, 100, 4, 13, Bd1);

  // embeddings
  {
    EmbNodeP P{hVs, hVv, leaf(L_WVLN_G), leaf(L_WVLN_B),
               leaf(L_WV_WH), leaf(L_WV_WSW), leaf(L_WV_WSB), leaf(L_WV_WV), s, v};
    k_node_embed<<<dim3(NN/16), dim3(32), 0, stream>>>(P);
  }
  {
    EmbEdgeP P{hEs, hEv, leaf(L_WELN_G), leaf(L_WELN_B),
               leaf(L_WE_WH), leaf(L_WE_WSW), leaf(L_WE_WSB), leaf(L_WE_WV), es, ev};
    k_edge_embed<<<dim3(NE/16), dim3(32), 0, stream>>>(P);
  }

  const int dhN = NN*NS + NN*NV*3 + NN;
  for (int L=0; L<3; ++L){
    int b = L_LAYER0 + 24*L;
    k_zero<<<dim3((dhN+255)/256), dim3(256), 0, stream>>>(dh, dhN);
    MsgP M;
    M.s=s; M.v=v; M.es=es; M.ev=ev; M.ei=ei;
    M.B0=B0[L]; M.B1=B1[L]; M.B2=B2[L];
    M.b0=leaf(b+9); M.b1=leaf(b+13); M.b2=leaf(b+17);
    M.wh0=leaf(b+8);  M.wv0=leaf(b+11);
    M.wh1=leaf(b+12); M.wv1=leaf(b+15);
    M.wh2=leaf(b+16); M.wv2=leaf(b+19);
    M.dh_s=dh_s; M.dh_v=dh_v; M.cnt=cnt;
    k_message<<<dim3(NE/32), dim3(64), 0, stream>>>(M);
    NodeP U;
    U.s=s; U.v=v; U.dh_s=dh_s; U.dh_v=dh_v; U.cnt=cnt;
    U.g0=leaf(b+21); U.bb0=leaf(b+20); U.g1=leaf(b+23); U.bb1=leaf(b+22);
    U.fwh0=leaf(b+0); U.fwv0=leaf(b+3); U.fwh1=leaf(b+4); U.fwv1=leaf(b+7);
    U.Bff0=Bff0[L]; U.Bff1=Bff1[L]; U.bff0=leaf(b+1); U.bff1=leaf(b+5);
    k_node_update<<<dim3(NN/16), dim3(32), 0, stream>>>(U);
  }

  {
    OutP P;
    P.s=s; P.v=v; P.gO=leaf(L_WOLN_G); P.bO=leaf(L_WOLN_B);
    P.whO=leaf(L_WO_WH); P.wsbO=leaf(L_WO_WSB);
    P.BO=BO; P.Bd1=Bd1; P.bd1=leaf(L_D1_B); P.wd2=leaf(L_D2_W); P.bd2=leaf(L_D2_B);
    P.o=o;
    k_out<<<dim3(NN/16), dim3(32), 0, stream>>>(P);
  }
  k_pairsum<<<dim3((NN/256)*NN), dim3(256), 0, stream>>>(o, (float*)d_out);
}
